// ByteEncoder_7172595384945
// MI455X (gfx1250) — matrix operand order repaired and checked by reading, not compiled, not run on hardware
//
#include <hip/hip_runtime.h>
#include <hip/hip_bf16.h>
#include <math.h>

#define B_ 4
#define S_ 2048
#define D_ 512
#define H_ 8
#define V_ 100000
#define P_ 256
#define DH_ 64
#define NG_ 6

typedef _Float16 v16h __attribute__((ext_vector_type(16)));
typedef _Float16 v4h  __attribute__((ext_vector_type(4)));
typedef float    v8f  __attribute__((ext_vector_type(8)));

__device__ __forceinline__ v8f wmma_f16(v16h a, v16h b, v8f c) {
  // D = A(16x32 f16) x B(32x16 f16) + C(16x16 f32)
  return __builtin_amdgcn_wmma_f32_16x16x32_f16(
      false, a, false, b, (short)0, c, false, false);
}

// Fragment load: lane L's 16 halves contiguous at base + L*16 (32B aligned)
// -> 2x ds_load_b128.
__device__ __forceinline__ v16h frag_ld(const _Float16* base) {
  int lane = threadIdx.x & 31;
  return *(const v16h*)(base + lane * 16);
}

// f16 WMMA operand k order used by this code, contiguous (inverse maps used when staging into LDS):
// A 16x32: lane = M + 16*(k>>4),     elem = k&15
// B 32x16: lane = N + 16*(k>>4),     elem = k&15
__device__ __forceinline__ int a_lane(int m, int k) { return m | ((k >> 4) << 4); }
__device__ __forceinline__ int a_elem(int k)        { return k & 15; }

__device__ __forceinline__ v4h cvt4(float4 f) {
  v4h h;
  h[0] = (_Float16)f.x; h[1] = (_Float16)f.y;
  h[2] = (_Float16)f.z; h[3] = (_Float16)f.w;
  return h;
}

// ---------------------------------------------------------------------------
// Embedding: byte table + 6 hashed n-gram tables, averaged by 1/7.
// 128 threads, one float4 (4 dims) per thread.
// ---------------------------------------------------------------------------
__global__ __launch_bounds__(128) void embed_kernel(
    const int* __restrict__ bytes, const float* __restrict__ byte_emb,
    const float* __restrict__ ngram_emb, float* __restrict__ E) {
  int tok = blockIdx.x;
  int b = tok / S_, s = tok % S_;
  const int* row = bytes + (long long)b * S_;
  int by[8];
#pragma unroll
  for (int k = 0; k < 8; ++k) {
    int idx = s - 7 + k;
    by[k] = (idx >= 0) ? row[idx] : 0;
  }
  long long hh[NG_];
#pragma unroll
  for (int j = 0; j < NG_; ++j) {
    int n = j + 3;
    unsigned long long h = 0ull;
    if (s >= n - 1) {
      for (int k = 0; k < n; ++k)
        h += (unsigned long long)(unsigned)by[8 - n + k] << (8 * k);
    }
    long long r = (long long)h % (long long)V_;   // int64 wraparound semantics
    if (r < 0) r += V_;
    hh[j] = r;
  }
  int d = threadIdx.x * 4;
  float4 acc = *(const float4*)(byte_emb + (long long)by[7] * D_ + d);
#pragma unroll
  for (int j = 0; j < NG_; ++j) {
    float4 gg = *(const float4*)(ngram_emb + ((long long)j * V_ + hh[j]) * D_ + d);
    acc.x += gg.x; acc.y += gg.y; acc.z += gg.z; acc.w += gg.w;
  }
  const float inv7 = 1.0f / 7.0f;
  float4 o; o.x = acc.x * inv7; o.y = acc.y * inv7;
  o.z = acc.z * inv7; o.w = acc.w * inv7;
  *(float4*)(E + (long long)tok * D_ + d) = o;
}

// ---------------------------------------------------------------------------
// WMMA GEMM: C[M,N] = A[M,K]*W[K,N] + bias, optional relu.
// 128 threads (4 waves), block tile 128x64, K-step 32, register-pipelined
// global->LDS staging. LDS tiles in fragment order.
// ---------------------------------------------------------------------------
__global__ __launch_bounds__(128) void gemm_kernel(
    const float* __restrict__ A, const float* __restrict__ W,
    const float* __restrict__ bias, float* __restrict__ C,
    int M, int N, int K, int relu) {
  __shared__ alignas(32) _Float16 Asf[8 * 512];   // 8 wave-tiles (16x32)
  __shared__ alignas(32) _Float16 Bsf[4 * 512];   // 4 n-tiles (32x16)
  int n0 = blockIdx.x * 64;
  int m0 = blockIdx.y * 128;
  int t = threadIdx.x;
  int w = t >> 5, lane = t & 31;

  // Static per-thread staging geometry.
  const int ar = t >> 3;             // A row within tile strip (m), strips e*16
  const int ak = (t & 7) * 4;        // A k (float4 aligned)
  const int bk = t >> 4;             // B k base, + e*8
  const int bn = (t & 15) * 4;       // B col base within 64
  const float* Abase = A + (long long)(m0 + ar) * K + ak;
  const float* Wbase = W + (long long)bk * N + n0 + bn;
  _Float16* AsDst = Asf + a_lane(ar, ak) * 16 + a_elem(ak);   // + e*512

  float4 aR[8], bR[4];
  const int nk = K / 32;

  auto preload = [&](int k0) {
#pragma unroll
    for (int e = 0; e < 8; ++e)
      aR[e] = *(const float4*)(Abase + (long long)e * 16 * K + k0);
#pragma unroll
    for (int e = 0; e < 4; ++e)
      bR[e] = *(const float4*)(Wbase + (long long)(k0 + e * 8) * N);
  };

  v8f zero = {};
  v8f acc[2][4];
#pragma unroll
  for (int s = 0; s < 2; ++s)
#pragma unroll
    for (int i = 0; i < 4; ++i) acc[s][i] = zero;

  preload(0);
  for (int it = 0; it < nk; ++it) {
    __syncthreads();                       // previous compute done with LDS
#pragma unroll
    for (int e = 0; e < 8; ++e)            // A: one ds_store_b64 per strip
      *(v4h*)(AsDst + e * 512) = cvt4(aR[e]);
#pragma unroll
    for (int e = 0; e < 4; ++e) {          // B: 4 scattered b16 per float4
      int k = bk + e * 8;
      int eb = k & 15, g = k >> 4;
      v4h hv = cvt4(bR[e]);
#pragma unroll
      for (int i = 0; i < 4; ++i) {
        int n = bn + i;
        Bsf[(n >> 4) * 512 + ((n & 15) | (g << 4)) * 16 + eb] = hv[i];
      }
    }
    __syncthreads();
    if (it + 1 < nk) {
      preload((it + 1) * 32);              // overlap next tile with WMMAs
      if (it + 2 < nk)
        __builtin_prefetch(Abase + (long long)(it + 2) * 32, 0, 0);
    }
    v16h af0 = frag_ld(Asf + (2 * w + 0) * 512);
    v16h af1 = frag_ld(Asf + (2 * w + 1) * 512);
#pragma unroll
    for (int nt = 0; nt < 4; ++nt) {
      v16h bf = frag_ld(Bsf + nt * 512);
      acc[0][nt] = wmma_f16(af0, bf, acc[0][nt]);
      acc[1][nt] = wmma_f16(af1, bf, acc[1][nt]);
    }
  }
  int g = lane >> 4, n = lane & 15;
#pragma unroll
  for (int s = 0; s < 2; ++s) {
#pragma unroll
    for (int nt = 0; nt < 4; ++nt) {
      int col = n0 + nt * 16 + n;
      float bv = bias[col];
#pragma unroll
      for (int r = 0; r < 8; ++r) {
        int rowm = m0 + w * 32 + s * 16 + r + 8 * g;
        float v = acc[s][nt][r] + bv;
        if (relu) v = fmaxf(v, 0.0f);
        C[(long long)rowm * N + col] = v;
      }
    }
  }
}

// ---------------------------------------------------------------------------
// Flash attention (online softmax), head dim 64, key blocks of 32.
// Grid: (Lq/64, H, B). 128 threads = 4 waves; wave owns 16 q-rows.
// Q/K/V are [B*L, D], head h at columns [h*64, h*64+64). Register-pipelined
// K/V staging; all LDS tiles in fragment order.
// ---------------------------------------------------------------------------
__global__ __launch_bounds__(128) void flash_kernel(
    const float* __restrict__ Qb, const float* __restrict__ Kb,
    const float* __restrict__ Vb, float* __restrict__ Ob,
    int Lq, int Lk) {
  __shared__ alignas(32) _Float16 Qf[4 * 2 * 512];  // 4 wave-tiles x 2 k-halves
  __shared__ alignas(32) _Float16 Kf[4 * 512];      // (kc*2+nt) tiles for Q.K^T
  __shared__ alignas(32) _Float16 Vf[4 * 512];      // 4 n-tiles for P.V
  __shared__ alignas(32) _Float16 Pf[4 * 512];      // per-wave P (A-frag order)
  int qb = blockIdx.x, h = blockIdx.y, b = blockIdx.z;
  int t = threadIdx.x, w = t >> 5, lane = t & 31;
  int g = lane >> 4, nl = lane & 15;

  // ---- Q tile (once): float4 loads, b64 frag-order stores -----------------
  {
    int qr = t >> 4;                 // row base, + e*8
    int d = (t & 15) * 4;
    int kh = d >> 5, kl = d & 31;
    const float* Qsrc = Qb + ((long long)b * Lq + qb * 64 + qr) * D_ + h * DH_ + d;
#pragma unroll
    for (int e = 0; e < 8; ++e) {
      int r = qr + e * 8;
      int wt = r >> 4, m = r & 15;
      float4 q4 = *(const float4*)(Qsrc + (long long)e * 8 * D_);
      *(v4h*)(Qf + wt * 1024 + kh * 512 + a_lane(m, kl) * 16 + a_elem(kl)) = cvt4(q4);
    }
  }
  __syncthreads();
  v16h aq0 = frag_ld(Qf + w * 1024);           // dh 0..31
  v16h aq1 = frag_ld(Qf + w * 1024 + 512);     // dh 32..63

  // ---- K/V staging geometry ----------------------------------------------
  const int kvr = t >> 4;            // key base, + e*8
  const int kvd = (t & 15) * 4;      // dh base (float4 aligned)
  const float* Ksrc = Kb + ((long long)b * Lk + kvr) * D_ + h * DH_ + kvd;
  const float* Vsrc = Vb + ((long long)b * Lk + kvr) * D_ + h * DH_ + kvd;
  const int kc = kvd >> 5, klg = (kvd >> 4) & 1, keb = kvd & 15;

  float4 kR[4], vR[4];
  auto preloadKV = [&](int kb0) {
#pragma unroll
    for (int e = 0; e < 4; ++e) {
      long long off = (long long)(kb0 + e * 8) * D_;
      kR[e] = *(const float4*)(Ksrc + off);
      vR[e] = *(const float4*)(Vsrc + off);
    }
  };

  float mrow[8], lrow[8];
  v8f zero = {};
  v8f oacc[4];
#pragma unroll
  for (int r = 0; r < 8; ++r) { mrow[r] = -1.0e30f; lrow[r] = 0.0f; }
#pragma unroll
  for (int i = 0; i < 4; ++i) oacc[i] = zero;
  const float scale = 0.125f;   // DH^-0.5

  preloadKV(0);
  for (int kb0 = 0; kb0 < Lk; kb0 += 32) {
    __syncthreads();
#pragma unroll
    for (int e = 0; e < 4; ++e) {            // K^T: contraction=dh -> b64 store
      int key = kvr + e * 8;
      *(v4h*)(Kf + (kc * 2 + (key >> 4)) * 512 +
              ((key & 15) | (klg << 4)) * 16 + keb) = cvt4(kR[e]);
    }
#pragma unroll
    for (int e = 0; e < 4; ++e) {            // V: contraction=key -> 4x b16
      int key = kvr + e * 8;
      int eb = key & 15, gk = key >> 4;
      v4h hv = cvt4(vR[e]);
#pragma unroll
      for (int i = 0; i < 4; ++i) {
        int d = kvd + i;
        Vf[(d >> 4) * 512 + ((d & 15) | (gk << 4)) * 16 + eb] = hv[i];
      }
    }
    __syncthreads();
    if (kb0 + 32 < Lk) preloadKV(kb0 + 32);  // overlap with compute below

    // S(16x32) = Q(16x64).K^T(64x32): two chained K=32 WMMAs per n-tile
    v8f s0 = zero, s1 = zero;
    s0 = wmma_f16(aq0, frag_ld(Kf + 0 * 512), s0);
    s0 = wmma_f16(aq1, frag_ld(Kf + 2 * 512), s0);
    s1 = wmma_f16(aq0, frag_ld(Kf + 1 * 512), s1);
    s1 = wmma_f16(aq1, frag_ld(Kf + 3 * 512), s1);

    _Float16* Pw = Pf + w * 512;
#pragma unroll
    for (int r = 0; r < 8; ++r) {            // online softmax per row r+8*g
      float a0 = s0[r] * scale, a1 = s1[r] * scale;
      float mx = fmaxf(a0, a1);
#pragma unroll
      for (int msk = 8; msk >= 1; msk >>= 1)
        mx = fmaxf(mx, __shfl_xor(mx, msk, 32));
      float mnew = fmaxf(mrow[r], mx);
      float alpha = __expf(mrow[r] - mnew);
      float p0 = __expf(a0 - mnew), p1 = __expf(a1 - mnew);
      float ps = p0 + p1;
#pragma unroll
      for (int msk = 8; msk >= 1; msk >>= 1)
        ps += __shfl_xor(ps, msk, 32);
      lrow[r] = lrow[r] * alpha + ps;
      mrow[r] = mnew;
#pragma unroll
      for (int nt = 0; nt < 4; ++nt)
        oacc[nt][r] = oacc[nt][r] * alpha;
      // Write P directly in A-frag order: row m=r+8g, keys nl and nl+16.
      int dlane = r + 8 * g;
      int off = nl;
      Pw[dlane * 16 + off]        = (_Float16)p0;
      Pw[(dlane + 16) * 16 + off] = (_Float16)p1;
    }
    // Per-wave LDS RAW: wait before re-reading P as an A-fragment.
    asm volatile("s_wait_dscnt 0" ::: "memory");
    v16h pa = frag_ld(Pw);
#pragma unroll
    for (int nt = 0; nt < 4; ++nt) {
      v16h vb = frag_ld(Vf + nt * 512);
      oacc[nt] = wmma_f16(pa, vb, oacc[nt]);   // O += P(16x32).V(32x64)
    }
  }

#pragma unroll
  for (int r = 0; r < 8; ++r) {
    float inv = 1.0f / lrow[r];
    long long orow = (long long)b * Lq + qb * 64 + w * 16 + r + 8 * g;
#pragma unroll
    for (int nt = 0; nt < 4; ++nt)
      Ob[orow * D_ + h * DH_ + nt * 16 + nl] = oacc[nt][r] * inv;
  }
}

// ---------------------------------------------------------------------------
// out = LayerNorm(X + R)*gamma + beta, one 512-wide row per block (float4).
// ---------------------------------------------------------------------------
__global__ __launch_bounds__(128) void add_ln_kernel(
    const float* __restrict__ X, const float* __restrict__ R,
    const float* __restrict__ gamma, const float* __restrict__ beta,
    float* __restrict__ O) {
  long long row = blockIdx.x;
  int t = threadIdx.x, d = t * 4;
  float4 x4 = *(const float4*)(X + row * D_ + d);
  float4 r4 = *(const float4*)(R + row * D_ + d);
  float v0 = x4.x + r4.x, v1 = x4.y + r4.y, v2 = x4.z + r4.z, v3 = x4.w + r4.w;
  float s = v0 + v1 + v2 + v3;
  float s2 = v0 * v0 + v1 * v1 + v2 * v2 + v3 * v3;
#pragma unroll
  for (int m = 16; m >= 1; m >>= 1) {
    s  += __shfl_xor(s,  m, 32);
    s2 += __shfl_xor(s2, m, 32);
  }
  __shared__ float red[8];
  int w = t >> 5, lane = t & 31;
  if (lane == 0) { red[w] = s; red[4 + w] = s2; }
  __syncthreads();
  s  = red[0] + red[1] + red[2] + red[3];
  s2 = red[4] + red[5] + red[6] + red[7];
  float mean = s * (1.0f / D_);
  float var  = s2 * (1.0f / D_) - mean * mean;
  float rstd = rsqrtf(var + 1e-5f);
  float4 g4 = *(const float4*)(gamma + d);
  float4 b4 = *(const float4*)(beta + d);
  float4 o;
  o.x = g4.x * (v0 - mean) * rstd + b4.x;
  o.y = g4.y * (v1 - mean) * rstd + b4.y;
  o.z = g4.z * (v2 - mean) * rstd + b4.z;
  o.w = g4.w * (v3 - mean) * rstd + b4.w;
  *(float4*)(O + row * D_ + d) = o;
}

// ---------------------------------------------------------------------------
// Gather patch rows: O[b*P+p, :] = X[b*S + idx[b,p], :]  (float4 copy)
// ---------------------------------------------------------------------------
__global__ __launch_bounds__(128) void gather_kernel(
    const float* __restrict__ X, const int* __restrict__ idx,
    float* __restrict__ O) {
  int bp = blockIdx.x;
  int b = bp / P_;
  long long src = (long long)b * S_ + idx[bp];
  int d = threadIdx.x * 4;
  *(float4*)(O + (long long)bp * D_ + d) = *(const float4*)(X + src * D_ + d);
}

// ---------------------------------------------------------------------------
extern "C" void kernel_launch(void* const* d_in, const int* in_sizes, int n_in,
                              void* d_out, int out_size, void* d_ws, size_t ws_size,
                              hipStream_t stream) {
  const int*   bytes     = (const int*)d_in[0];
  const int*   patch_idx = (const int*)d_in[1];
  const float* byte_emb  = (const float*)d_in[2];
  const float* ngram_emb = (const float*)d_in[3];
  const float* sWq = (const float*)d_in[4];  const float* sbq = (const float*)d_in[5];
  const float* sWk = (const float*)d_in[6];  const float* sbk = (const float*)d_in[7];
  const float* sWv = (const float*)d_in[8];  const float* sbv = (const float*)d_in[9];
  const float* sWo = (const float*)d_in[10]; const float* sbo = (const float*)d_in[11];
  const float* ln1g = (const float*)d_in[12]; const float* ln1b = (const float*)d_in[13];
  const float* W1  = (const float*)d_in[14]; const float* b1  = (const float*)d_in[15];
  const float* W2  = (const float*)d_in[16]; const float* b2  = (const float*)d_in[17];
  const float* ln2g = (const float*)d_in[18]; const float* ln2b = (const float*)d_in[19];
  const float* cWq = (const float*)d_in[20]; const float* cbq = (const float*)d_in[21];
  const float* cWk = (const float*)d_in[22]; const float* cbk = (const float*)d_in[23];
  const float* cWv = (const float*)d_in[24]; const float* cbv = (const float*)d_in[25];
  const float* cWo = (const float*)d_in[26]; const float* cbo = (const float*)d_in[27];

  float* ws = (float*)d_ws;
  float* out = (float*)d_out;

  const long long MD = (long long)B_ * S_ * D_;     // 4,194,304 floats
  const long long PD = (long long)B_ * P_ * D_;     //   524,288 floats
  // Arena with reuse; peak = 7*MD floats (~117 MB).
  float* E  = ws;                 // embeddings        [8192,512]
  float* Qb = ws + MD;            // self Q
  float* Kb = ws + 2 * MD;        // self K
  float* Vb = ws + 3 * MD;        // self V
  float* Ab = ws + 4 * MD;        // self-attn output
  float* AO = Qb;                 // O-proj output  (Q dead after flash)
  float* X1 = Kb;                 // LN1 output     (K dead after flash)
  float* F1 = ws + 3 * MD;        // FFN hidden [8192,2048] (V,Ab dead)
  float* F2 = ws;                 // FFN out        (E dead after LN1)
  float* X2 = Qb;                 // LN2 output     (AO dead after LN1)
  float* Qg = ws;                 // gathered queries (F2 dead after LN2)
  float* cQ = ws + PD;            // cross Q [1024,512]
  float* cK = ws + 2 * MD;        // cross K (X1 dead after LN2)
  float* cV = ws + 3 * MD;        // cross V (F1 dead after FFN2)
  float* CA = ws + 2 * PD;        // cross-attn output [1024,512]

  dim3 blk(128);
  dim3 gemmBS(D_ / 64, (B_ * S_) / 128);       // N=512,  M=8192
  dim3 gemmFF1(2048 / 64, (B_ * S_) / 128);    // N=2048, M=8192
  dim3 gemmBP(D_ / 64, (B_ * P_) / 128);       // N=512,  M=1024

  embed_kernel<<<B_ * S_, blk, 0, stream>>>(bytes, byte_emb, ngram_emb, E);

  gemm_kernel<<<gemmBS, blk, 0, stream>>>(E, sWq, sbq, Qb, B_ * S_, D_, D_, 0);
  gemm_kernel<<<gemmBS, blk, 0, stream>>>(E, sWk, sbk, Kb, B_ * S_, D_, D_, 0);
  gemm_kernel<<<gemmBS, blk, 0, stream>>>(E, sWv, sbv, Vb, B_ * S_, D_, D_, 0);

  flash_kernel<<<dim3(S_ / 64, H_, B_), blk, 0, stream>>>(Qb, Kb, Vb, Ab, S_, S_);

  gemm_kernel<<<gemmBS, blk, 0, stream>>>(Ab, sWo, sbo, AO, B_ * S_, D_, D_, 0);
  add_ln_kernel<<<B_ * S_, blk, 0, stream>>>(E, AO, ln1g, ln1b, X1);

  gemm_kernel<<<gemmFF1, blk, 0, stream>>>(X1, W1, b1, F1, B_ * S_, 2048, D_, 1);
  gemm_kernel<<<gemmBS, blk, 0, stream>>>(F1, W2, b2, F2, B_ * S_, D_, 2048, 0);
  add_ln_kernel<<<B_ * S_, blk, 0, stream>>>(X1, F2, ln2g, ln2b, X2);

  gather_kernel<<<B_ * P_, blk, 0, stream>>>(X2, patch_idx, Qg);
  gemm_kernel<<<gemmBP, blk, 0, stream>>>(Qg, cWq, cbq, cQ, B_ * P_, D_, D_, 0);
  gemm_kernel<<<gemmBS, blk, 0, stream>>>(X2, cWk, cbk, cK, B_ * S_, D_, D_, 0);
  gemm_kernel<<<gemmBS, blk, 0, stream>>>(X2, cWv, cbv, cV, B_ * S_, D_, D_, 0);

  flash_kernel<<<dim3(P_ / 64, H_, B_), blk, 0, stream>>>(cQ, cK, cV, CA, P_, S_);

  gemm_kernel<<<gemmBP, blk, 0, stream>>>(CA, cWo, cbo, out, B_ * P_, D_, D_, 0);
}
